// NeighborhoodSelfAttentionBlock_1726576854078
// MI455X (gfx1250) — matrix operand order repaired and checked by reading, not compiled, not run on hardware
//
#include <hip/hip_runtime.h>
#include <hip/hip_bf16.h>

// ---------------------------------------------------------------------------
// NeighborhoodSelfAttentionBlock for MI455X (gfx1250, wave32, WMMA)
//   x:(4,96,96,512) f32, 7x7 NA, 8 heads x 64, RMSNorm(ada) -> QKV GEMM ->
//   cos-sim scale + RoPE -> NA2D softmax -> out GEMM + skip
// GEMMs: v_wmma_f32_16x16x32_bf16, 2x2 register-blocked fragments per wave
// (32x32 wave tile, 128x64 block tile). k-loop software pipelined with a
// PEELED epilogue (no conditional defs in the loop -> clean renaming).
// Attention: wave-per-(pixel,head) online softmax with shfl butterflies.
// ---------------------------------------------------------------------------

#define NB    4
#define HH    96
#define WW    96
#define DM    512
#define NH    8
#define DH    64
#define KS    7
#define NPIX  (NB * HH * WW)          // 36864
#define NQKV  (3 * DM)                // 1536
#define EPSF  1e-6f

typedef __attribute__((ext_vector_type(16))) __bf16 v16bf;
typedef __attribute__((ext_vector_type(8)))  float  v8f;

union BfPack { uint4 u[2]; v16bf v; };

// float -> bf16 bits, round-to-nearest-even
__device__ __forceinline__ unsigned short f2bf(float f) {
  unsigned int u = __float_as_uint(f);
  unsigned int r = u + 0x7FFFu + ((u >> 16) & 1u);
  return (unsigned short)(r >> 16);
}

__device__ __forceinline__ float wave_sum32(float v) {
  v += __shfl_xor(v, 16, 32);
  v += __shfl_xor(v, 8, 32);
  v += __shfl_xor(v, 4, 32);
  v += __shfl_xor(v, 2, 32);
  v += __shfl_xor(v, 1, 32);
  return v;
}

// --------------------------- kernel 1: ada ---------------------------------
// ada[b][j] = dot(cond[b,:], norm_w[j,:]) + 1
__global__ void k_ada(const float* __restrict__ cond,
                      const float* __restrict__ norm_w,
                      float* __restrict__ ada) {
  int tid = blockIdx.x * blockDim.x + threadIdx.x;
  if (tid >= NB * DM) return;
  int b = tid / DM, j = tid % DM;
  const float4* c4 = (const float4*)(cond + (size_t)b * DM);
  const float4* w4 = (const float4*)(norm_w + (size_t)j * DM);
  float acc = 0.f;
  for (int k = 0; k < DM / 4; ++k) {
    float4 c = c4[k], w = w4[k];
    acc += c.x * w.x + c.y * w.y + c.z * w.z + c.w * w.w;
  }
  ada[tid] = acc + 1.0f;
}

// ----------------------- kernel 2: weight -> bf16 --------------------------
__global__ void k_cvt_w(const float* __restrict__ qkv_w,
                        const float* __restrict__ out_w,
                        unsigned short* __restrict__ qkvw_bf,
                        unsigned short* __restrict__ outw_bf) {
  int tid = blockIdx.x * blockDim.x + threadIdx.x;
  const int nq = NQKV * DM;           // 786432
  const int no = DM * DM;             // 262144
  if (tid < nq) qkvw_bf[tid] = f2bf(qkv_w[tid]);
  else if (tid < nq + no) outw_bf[tid - nq] = f2bf(out_w[tid - nq]);
}

// ---------------- kernel 3: RMSNorm * ada -> bf16 activations --------------
// one block (128 thr) per pixel row of 512
__global__ void k_rmsnorm(const float* __restrict__ x,
                          const float* __restrict__ ada,
                          unsigned short* __restrict__ xn_bf) {
  __shared__ float red[4];
  int pix = blockIdx.x;
  int t = threadIdx.x;                 // 0..127
  int lane = t & 31, wid = t >> 5;
  int b = pix / (HH * WW);
  const float4* xr = (const float4*)(x + (size_t)pix * DM);
  float4 v = xr[t];
  float ss = v.x * v.x + v.y * v.y + v.z * v.z + v.w * v.w;
  ss = wave_sum32(ss);
  if (lane == 0) red[wid] = ss;
  __syncthreads();
  float tot = red[0] + red[1] + red[2] + red[3];
  float inv = rsqrtf(tot * (1.0f / DM) + EPSF);
  const float* ar = ada + (size_t)b * DM;
  int j = t * 4;
  float n0 = v.x * inv * ar[j + 0];
  float n1 = v.y * inv * ar[j + 1];
  float n2 = v.z * inv * ar[j + 2];
  float n3 = v.w * inv * ar[j + 3];
  unsigned int* dst = (unsigned int*)(xn_bf + (size_t)pix * DM + j);
  dst[0] = (unsigned int)f2bf(n0) | ((unsigned int)f2bf(n1) << 16);
  dst[1] = (unsigned int)f2bf(n2) | ((unsigned int)f2bf(n3) << 16);
}

// ------------------- kernels 4/7: WMMA bf16 GEMM ---------------------------
// C[M,N] = A[M,K] * B[N,K]^T  (+ skip)    A,B are bf16 bits in ushort
// block = 256 thr = 8 waves (4M x 2N); wave computes a 32x32 tile via a
// 2x2 grid of 16x16 WMMA fragments. k-loop software pipelined with peeled
// epilogue: loop body loads step k+32 unconditionally, WMMAs consume step k.
template <bool ADD_SKIP>
__global__ void k_gemm_bf16(const unsigned short* __restrict__ A,
                            const unsigned short* __restrict__ B,
                            float* __restrict__ C,
                            const float* __restrict__ skip,
                            int M, int N, int K) {
  int wave = threadIdx.x >> 5;
  int lane = threadIdx.x & 31;
  int l = lane & 15, hi = lane >> 4;
  int tile_m = blockIdx.x * 128 + (wave >> 1) * 32;
  int tile_n = blockIdx.y * 64 + (wave & 1) * 32;

  // A 16x32 bf16 frag: VGPR0-3 = K[k0+hi*8 .. +7], VGPR4-7 = K[k0+16+hi*8 .. +7]
  const unsigned short* arow0 = A + (size_t)(tile_m + l) * K + hi * 8;
  const unsigned short* arow1 = A + (size_t)(tile_m + 16 + l) * K + hi * 8;
  // B 32x16 bf16 frag (col n = B-row n in memory): VGPR0-3 = K[k0+hi*8 .. +7], VGPR4-7 = K[k0+16+hi*8 .. +7]
  const unsigned short* brow0 = B + (size_t)(tile_n + l) * K + hi * 8;
  const unsigned short* brow1 = B + (size_t)(tile_n + 16 + l) * K + hi * 8;

  v8f acc00 = {}, acc01 = {}, acc10 = {}, acc11 = {};
  BfPack a0, a1, b0, b1;

  // prologue: load k-step 0
  a0.u[0] = *(const uint4*)(arow0);
  a0.u[1] = *(const uint4*)(arow0 + 16);
  a1.u[0] = *(const uint4*)(arow1);
  a1.u[1] = *(const uint4*)(arow1 + 16);
  b0.u[0] = *(const uint4*)(brow0);
  b0.u[1] = *(const uint4*)(brow0 + 16);
  b1.u[0] = *(const uint4*)(brow1);
  b1.u[1] = *(const uint4*)(brow1 + 16);

  // steady state: unconditional next-step loads, then WMMAs on current step
#pragma unroll 2
  for (int kn = 32; kn < K; kn += 32) {
    BfPack na0, na1, nb0, nb1;
    na0.u[0] = *(const uint4*)(arow0 + kn);
    na0.u[1] = *(const uint4*)(arow0 + kn + 16);
    na1.u[0] = *(const uint4*)(arow1 + kn);
    na1.u[1] = *(const uint4*)(arow1 + kn + 16);
    nb0.u[0] = *(const uint4*)(brow0 + kn);
    nb0.u[1] = *(const uint4*)(brow0 + kn + 16);
    nb1.u[0] = *(const uint4*)(brow1 + kn);
    nb1.u[1] = *(const uint4*)(brow1 + kn + 16);
    if (kn + 32 < K) {
      __builtin_prefetch(arow0 + kn + 32, 0, 1);  // global_prefetch_b8
      __builtin_prefetch(arow1 + kn + 32, 0, 1);
      __builtin_prefetch(brow0 + kn + 32, 0, 1);
      __builtin_prefetch(brow1 + kn + 32, 0, 1);
    }
    acc00 = __builtin_amdgcn_wmma_f32_16x16x32_bf16(
        false, a0.v, false, b0.v, (short)0, acc00, false, false);
    acc01 = __builtin_amdgcn_wmma_f32_16x16x32_bf16(
        false, a0.v, false, b1.v, (short)0, acc01, false, false);
    acc10 = __builtin_amdgcn_wmma_f32_16x16x32_bf16(
        false, a1.v, false, b0.v, (short)0, acc10, false, false);
    acc11 = __builtin_amdgcn_wmma_f32_16x16x32_bf16(
        false, a1.v, false, b1.v, (short)0, acc11, false, false);
    a0 = na0; a1 = na1; b0 = nb0; b1 = nb1;   // renamed away (unconditional defs)
  }

  // peeled epilogue: WMMAs on the final k-step
  acc00 = __builtin_amdgcn_wmma_f32_16x16x32_bf16(
      false, a0.v, false, b0.v, (short)0, acc00, false, false);
  acc01 = __builtin_amdgcn_wmma_f32_16x16x32_bf16(
      false, a0.v, false, b1.v, (short)0, acc01, false, false);
  acc10 = __builtin_amdgcn_wmma_f32_16x16x32_bf16(
      false, a1.v, false, b0.v, (short)0, acc10, false, false);
  acc11 = __builtin_amdgcn_wmma_f32_16x16x32_bf16(
      false, a1.v, false, b1.v, (short)0, acc11, false, false);

  // D 16x16 f32: VGPR r -> row (r + hi*8), col l   (per fragment)
#pragma unroll
  for (int fm = 0; fm < 2; ++fm) {
#pragma unroll
    for (int fn = 0; fn < 2; ++fn) {
      v8f acc = (fm == 0) ? ((fn == 0) ? acc00 : acc01)
                          : ((fn == 0) ? acc10 : acc11);
      size_t base = (size_t)(tile_m + fm * 16 + hi * 8) * N +
                    (tile_n + fn * 16 + l);
#pragma unroll
      for (int r = 0; r < 8; ++r) {
        size_t idx = base + (size_t)r * N;
        float val = acc[r];
        if (ADD_SKIP) val += skip[idx];
        C[idx] = val;
      }
    }
  }
}

// ------------- kernel 5: cosine-sim scale + RoPE on q,k (in place) ---------
// one wave per (pixel, head); lane owns dims (2l, 2l+1)
__global__ void k_qk_post(float* __restrict__ qkv,
                          const float* __restrict__ pos,
                          const float* __restrict__ scale) {
  const float LOGMIN = 1.1447298858494002f;            // ln(pi)
  const float STEP   = 0.03597789207805045f;           // ln(10)/64
  int unit = blockIdx.x * 8 + (threadIdx.x >> 5);
  int lane = threadIdx.x & 31;
  int pix = unit >> 3, h = unit & 7;

  float ph = pos[(size_t)pix * 2 + 0];
  float pw = pos[(size_t)pix * 2 + 1];
  float ss = sqrtf(scale[h]);

  // thetas for lanes < 16 (dims 2l,2l+1 < 32)
  int d0 = 2 * lane;
  int jj0 = d0 & 15, jj1 = jj0 + 1;
  float t0 = ((jj0 < 8) ? ph : pw) * __expf(LOGMIN + (float)((jj0 & 7) * 8 + h) * STEP);
  float t1 = ((jj1 < 8) ? ph : pw) * __expf(LOGMIN + (float)((jj1 & 7) * 8 + h) * STEP);
  float c0 = __cosf(t0), s0 = __sinf(t0);
  float c1 = __cosf(t1), s1 = __sinf(t1);

#pragma unroll
  for (int which = 0; which < 2; ++which) {            // q then k
    float2* p = (float2*)(qkv + (size_t)pix * NQKV + which * DM + h * DH);
    float2 v = p[lane];
    float sq = wave_sum32(v.x * v.x + v.y * v.y);      // sum over 64 dims
    float sc = ss * rsqrtf(sq + EPSF);
    v.x *= sc; v.y *= sc;
    float ox = __shfl_xor(v.x, 8, 32);                 // partner half
    float oy = __shfl_xor(v.y, 8, 32);
    if (lane < 16) {
      if (lane < 8) {                                  // x1' = x1*c - x2*s
        v.x = v.x * c0 - ox * s0;
        v.y = v.y * c1 - oy * s1;
      } else {                                         // x2' = x2*c + x1*s
        v.x = v.x * c0 + ox * s0;
        v.y = v.y * c1 + oy * s1;
      }
    }
    p[lane] = v;
  }
}

// -------- kernel 6: NA2D 7x7 attention, online softmax, -> bf16 ------------
// one wave per (pixel, head); lane owns out dims (2l, 2l+1)
__global__ void k_na2d(const float* __restrict__ qkv,
                       unsigned short* __restrict__ attn_bf) {
  int unit = blockIdx.x * 8 + (threadIdx.x >> 5);
  int lane = threadIdx.x & 31;
  int pix = unit >> 3, h = unit & 7;
  int b = pix / (HH * WW);
  int rem = pix - b * (HH * WW);
  int i = rem / WW, j = rem - i * WW;
  int sh = min(max(i - KS / 2, 0), HH - KS);
  int sw = min(max(j - KS / 2, 0), WW - KS);

  const float2* qp = (const float2*)(qkv + (size_t)pix * NQKV + 0 * DM + h * DH);
  float2 q = qp[lane];

  float m = -1e30f, sum = 0.f, a0 = 0.f, a1 = 0.f;
#pragma unroll 1
  for (int a = 0; a < KS; ++a) {
    int rowpix = (b * HH + sh + a) * WW + sw;
#pragma unroll 1
    for (int c = 0; c < KS; ++c) {
      size_t kp = (size_t)(rowpix + c);
      const float2* kk = (const float2*)(qkv + kp * NQKV + 1 * DM + h * DH);
      const float2* vv = (const float2*)(qkv + kp * NQKV + 2 * DM + h * DH);
      float2 kv = kk[lane];
      float s = wave_sum32(q.x * kv.x + q.y * kv.y);   // score on all lanes
      float2 vvl = vv[lane];
      float mn = fmaxf(m, s);
      float corr = __expf(m - mn);
      float p = __expf(s - mn);
      sum = sum * corr + p;
      a0 = a0 * corr + p * vvl.x;
      a1 = a1 * corr + p * vvl.y;
      m = mn;
    }
  }
  float inv = 1.0f / sum;
  unsigned int* dst = (unsigned int*)(attn_bf + (size_t)pix * DM + h * DH);
  dst[lane] = (unsigned int)f2bf(a0 * inv) | ((unsigned int)f2bf(a1 * inv) << 16);
}

// ---------------------------------------------------------------------------
extern "C" void kernel_launch(void* const* d_in, const int* in_sizes, int n_in,
                              void* d_out, int out_size, void* d_ws, size_t ws_size,
                              hipStream_t stream) {
  const float* x      = (const float*)d_in[0];   // (4,96,96,512)
  const float* pos    = (const float*)d_in[1];   // (4,96,96,2)
  const float* cond   = (const float*)d_in[2];   // (4,512)
  const float* norm_w = (const float*)d_in[3];   // (512,512)
  const float* qkv_w  = (const float*)d_in[4];   // (1536,512)
  const float* scale  = (const float*)d_in[5];   // (8,)
  const float* out_w  = (const float*)d_in[6];   // (512,512)
  float* out = (float*)d_out;

  char* ws = (char*)d_ws;
  size_t off = 0;
  float*          ada      = (float*)(ws + off);          off += (size_t)NB * DM * 4;                 // 8 KB
  unsigned short* qkvw_bf  = (unsigned short*)(ws + off); off += (size_t)NQKV * DM * 2;               // 1.5 MB
  unsigned short* outw_bf  = (unsigned short*)(ws + off); off += (size_t)DM * DM * 2;                 // 0.5 MB
  unsigned short* xn_bf    = (unsigned short*)(ws + off); off += (size_t)NPIX * DM * 2;               // 36 MB
  float*          qkv      = (float*)(ws + off);          off += (size_t)NPIX * NQKV * 4;             // 216 MB
  unsigned short* attn_bf  = (unsigned short*)(ws + off); off += (size_t)NPIX * DM * 2;               // 36 MB
  (void)ws_size; (void)in_sizes; (void)n_in; (void)out_size;

  // 1) ada
  k_ada<<<dim3((NB * DM + 255) / 256), dim3(256), 0, stream>>>(cond, norm_w, ada);
  // 2) weights -> bf16
  k_cvt_w<<<dim3((NQKV * DM + DM * DM + 255) / 256), dim3(256), 0, stream>>>(
      qkv_w, out_w, qkvw_bf, outw_bf);
  // 3) rmsnorm * ada -> bf16
  k_rmsnorm<<<dim3(NPIX), dim3(128), 0, stream>>>(x, ada, xn_bf);
  // 4) qkv = xn @ qkv_w^T   (WMMA bf16, 128x64 block tile)
  k_gemm_bf16<false><<<dim3(NPIX / 128, NQKV / 64), dim3(256), 0, stream>>>(
      xn_bf, qkvw_bf, qkv, nullptr, NPIX, NQKV, DM);
  // 5) cosine-sim scale + RoPE on q,k
  k_qk_post<<<dim3(NPIX), dim3(256), 0, stream>>>(qkv, pos, scale);
  // 6) neighborhood attention -> bf16
  k_na2d<<<dim3(NPIX), dim3(256), 0, stream>>>(qkv, attn_bf);
  // 7) out = attn @ out_w^T + x   (WMMA bf16 + skip epilogue)
  k_gemm_bf16<true><<<dim3(NPIX / 128, DM / 64), dim3(256), 0, stream>>>(
      attn_bf, outw_bf, out, x, NPIX, DM, DM);
}
